// Self_Attention_18949395710130
// MI455X (gfx1250) — matrix operand order repaired and checked by reading, not compiled, not run on hardware
//
#include <hip/hip_runtime.h>

typedef __attribute__((ext_vector_type(16))) _Float16 v16h;
typedef __attribute__((ext_vector_type(8)))  _Float16 v8h;
typedef __attribute__((ext_vector_type(8)))  float    v8f;

constexpr int BATCH = 8;
constexpr int CH    = 256;   // channels
constexpr int NPIX  = 4096;  // H*W
constexpr int DQK   = 32;    // q/k head dim == WMMA K
constexpr int NB    = 64;    // n-columns per flash phase

__device__ __forceinline__ v16h cat8(v8h lo, v8h hi) {
  return __builtin_shufflevector(lo, hi, 0,1,2,3,4,5,6,7,8,9,10,11,12,13,14,15);
}

// 16-lane max reduction via DPP16 butterflies (no LDS, no DScnt stalls).
template<int CTRL>
__device__ __forceinline__ float dppmax(float v) {
  int s = __builtin_amdgcn_update_dpp(0, __float_as_int(v), CTRL, 0xF, 0xF, true);
  return fmaxf(v, __int_as_float(s));
}
__device__ __forceinline__ float rowmax16(float v) {
  v = dppmax<0xB1>(v);   // quad_perm(1,0,3,2): xor 1
  v = dppmax<0x4E>(v);   // quad_perm(2,3,0,1): xor 2
  v = dppmax<0x141>(v);  // row_half_mirror  : combine across 4-groups
  v = dppmax<0x140>(v);  // row_mirror       : combine across 8-groups
  return v;
}

// ---------------------------------------------------------------------------
// Projection: out[d, n] = sum_c W[d,c] * x[b,c,n] + bias[d], cast to f16.
// PIXEL_MAJOR=true  -> out[b][n][D]   PIXEL_MAJOR=false -> out[b][d][N]
// ---------------------------------------------------------------------------
template<bool PIXEL_MAJOR>
__global__ __launch_bounds__(256) void proj_kernel(
    const float* __restrict__ W, const float* __restrict__ bias,
    const float* __restrict__ x, _Float16* __restrict__ out, int D)
{
  const int n = blockIdx.x * blockDim.x + threadIdx.x;
  const int d = blockIdx.y;
  const int b = blockIdx.z;
  const float* xb = x + ((size_t)b * CH) * NPIX + n;
  const float* wr = W + (size_t)d * CH;
  float acc = bias[d];
  #pragma unroll 8
  for (int c = 0; c < CH; ++c)
    acc = fmaf(wr[c], xb[(size_t)c * NPIX], acc);
  if (PIXEL_MAJOR)
    out[((size_t)(b * NPIX + n)) * D + d] = (_Float16)acc;
  else
    out[((size_t)(b * D + d)) * NPIX + n] = (_Float16)acc;
}

// ---------------------------------------------------------------------------
// Flash attention, WMMA f32_16x16x32_f16, barrier-free per-wave tiles.
// Block = 128 threads (4 waves); block owns (b, 16-row m-tile); each wave
// owns 4 of 16 output channel tiles, 64 n-columns per phase, loop unrolled
// 2x with ping-pong k-fragment buffers (no rotation movs).
// ---------------------------------------------------------------------------
__global__ __launch_bounds__(128) void attn_flash_wmma(
    const _Float16* __restrict__ qf,   // [B][N][32]
    const _Float16* __restrict__ kf,   // [B][N][32]
    const _Float16* __restrict__ vf,   // [B][C][N]
    const float*    __restrict__ x,    // [B][C][N]
    const float*    __restrict__ gamma,
    float*          __restrict__ out)  // [B][C][N]
{
  __shared__ _Float16 Pt[4][16][NB];   // per-wave P tile staging (8 KB total)

  const int wave = threadIdx.x >> 5;
  const int lane = threadIdx.x & 31;
  const int lh   = lane >> 4;          // half of wave (0/1)
  const int ln   = lane & 15;
  const int b    = blockIdx.y;
  const int m0   = blockIdx.x * 16;
  const int cb   = wave * 64;          // this wave's channel base (4 c-tiles)

  // q A-fragment: lane row m = m0+ln; elem j -> d = 16*lh + j (contiguous halves)
  const _Float16* qp = qf + ((size_t)(b * NPIX + m0 + ln)) * DQK;
  const v16h qa = cat8(*(const v8h*)(qp + 16 * lh),
                       *(const v8h*)(qp + 16 * lh + 8));

  // all-ones B fragment: rowsum(P) = P x ones
  v16h ones;
  #pragma unroll
  for (int j = 0; j < 16; ++j) ones[j] = (_Float16)1.0f;

  float Mr[8], Lr[8];                  // running row max / sum (row = r + 8*lh)
  #pragma unroll
  for (int r = 0; r < 8; ++r) { Mr[r] = -1e30f; Lr[r] = 0.0f; }

  v8f O[4];
  #pragma unroll
  for (int t = 0; t < 4; ++t)
    #pragma unroll
    for (int r = 0; r < 8; ++r) O[t][r] = 0.0f;

  v8f zacc;
  #pragma unroll
  for (int r = 0; r < 8; ++r) zacc[r] = 0.0f;

  // k B-fragment: col n ; elem j -> d = 16*lh + j (contiguous halves)
  const _Float16* kbase = kf + ((size_t)(b * NPIX + ln)) * DQK + 16 * lh;
  const _Float16* vbase = vf + ((size_t)(b * CH + cb + ln)) * NPIX + 16 * lh;

  // one flash phase: consume kbc for columns [n0, n0+NB); load kbnx for nload
  auto flash_phase = [&](int n0, const v16h* kbc, v16h* kbnx, int nload) {
    // ---- software pipeline: issue next phase's k loads first -------------
    if (nload < NPIX) {
      #pragma unroll
      for (int u = 0; u < 4; ++u) {
        const _Float16* kp = kbase + (size_t)(nload + 16 * u) * DQK;
        kbnx[u] = cat8(*(const v8h*)kp, *(const v8h*)(kp + 8));
      }
      if (nload + NB < NPIX)
        __builtin_prefetch(kbase + (size_t)(nload + NB) * DQK, 0, 1);
    }

    // ---- scores: four 16x16 tiles covering n0..n0+63 ---------------------
    v8f S[4];
    #pragma unroll
    for (int u = 0; u < 4; ++u)
      S[u] = __builtin_amdgcn_wmma_f32_16x16x32_f16(false, qa, false, kbc[u],
                                                    (short)0, zacc, false, false);

    // ---- online softmax: DPP row max, rescale, P tile to LDS -------------
    float alv[8];
    #pragma unroll
    for (int r = 0; r < 8; ++r) {
      float mx = fmaxf(fmaxf(S[0][r], S[1][r]), fmaxf(S[2][r], S[3][r]));
      mx = rowmax16(mx);                               // VALU-only reduction
      const float newM = fmaxf(Mr[r], mx);
      const float al   = __expf(Mr[r] - newM);         // 0 on first block
      alv[r] = al;
      Mr[r]  = newM;
      #pragma unroll
      for (int t = 0; t < 4; ++t) O[t][r] *= al;       // flash rescale
      #pragma unroll
      for (int u = 0; u < 4; ++u)
        Pt[wave][r + 8 * lh][16 * u + ln] = (_Float16)__expf(S[u][r] - newM);
    }

    // ---- P A-fragments from LDS (row = ln, K runs per 32-col chunk) ------
    const _Float16* prow = &Pt[wave][ln][0];
    const v16h pa0 = cat8(*(const v8h*)(prow + 16 * lh),
                          *(const v8h*)(prow + 8 + 16 * lh));
    const v16h pa1 = cat8(*(const v8h*)(prow + 32 + 16 * lh),
                          *(const v8h*)(prow + 40 + 16 * lh));

    // ---- row sums via WMMA: P x ones (lands in running-state layout) -----
    v8f Ls = __builtin_amdgcn_wmma_f32_16x16x32_f16(false, pa0, false, ones,
                                                    (short)0, zacc, false, false);
    Ls     = __builtin_amdgcn_wmma_f32_16x16x32_f16(false, pa1, false, ones,
                                                    (short)0, Ls, false, false);
    #pragma unroll
    for (int r = 0; r < 8; ++r) Lr[r] = Lr[r] * alv[r] + Ls[r];

    // ---- O[c-tile] += P x Vt, two K=32 chunks per c-tile -----------------
    #pragma unroll
    for (int t = 0; t < 4; ++t) {
      const _Float16* vp = vbase + (size_t)(16 * t) * NPIX + n0;
      if (n0 + NB < NPIX) __builtin_prefetch(vp + NB, 0, 1);
      const v16h vb0 = cat8(*(const v8h*)vp,        *(const v8h*)(vp + 8));
      const v16h vb1 = cat8(*(const v8h*)(vp + 32), *(const v8h*)(vp + 40));
      O[t] = __builtin_amdgcn_wmma_f32_16x16x32_f16(false, pa0, false, vb0,
                                                    (short)0, O[t], false, false);
      O[t] = __builtin_amdgcn_wmma_f32_16x16x32_f16(false, pa1, false, vb1,
                                                    (short)0, O[t], false, false);
    }
  };

  // ---- prologue: k fragments for block 0 ---------------------------------
  v16h kbA[4], kbB[4];
  #pragma unroll
  for (int u = 0; u < 4; ++u) {
    const _Float16* kp = kbase + (size_t)(16 * u) * DQK;
    kbA[u] = cat8(*(const v8h*)kp, *(const v8h*)(kp + 8));
  }

  // ---- main loop, 2x unrolled with ping-pong buffers ---------------------
  for (int n0 = 0; n0 < NPIX; n0 += 2 * NB) {
    flash_phase(n0,      kbA, kbB, n0 + NB);      // consume A, fill B
    flash_phase(n0 + NB, kbB, kbA, n0 + 2 * NB);  // consume B, fill A
  }

  // ---- epilogue: out = gamma * O/L + x (C/D layout: col=c, row=r+8*lh) ---
  const float g = gamma[0];
  float invL[8];
  #pragma unroll
  for (int r = 0; r < 8; ++r) invL[r] = 1.0f / Lr[r];
  #pragma unroll
  for (int t = 0; t < 4; ++t) {
    const int c = cb + 16 * t + ln;
    #pragma unroll
    for (int r = 0; r < 8; ++r) {
      const int m = m0 + r + 8 * lh;
      const size_t idx = ((size_t)(b * CH + c)) * NPIX + m;
      out[idx] = fmaf(g, O[t][r] * invL[r], x[idx]);
    }
  }
}

// ---------------------------------------------------------------------------
extern "C" void kernel_launch(void* const* d_in, const int* in_sizes, int n_in,
                              void* d_out, int out_size, void* d_ws, size_t ws_size,
                              hipStream_t stream) {
  (void)in_sizes; (void)n_in; (void)out_size; (void)ws_size;
  const float* x     = (const float*)d_in[0];
  const float* Wq    = (const float*)d_in[1];
  const float* bq    = (const float*)d_in[2];
  const float* Wk    = (const float*)d_in[3];
  const float* bk    = (const float*)d_in[4];
  const float* Wv    = (const float*)d_in[5];
  const float* bv    = (const float*)d_in[6];
  const float* gamma = (const float*)d_in[7];
  float* out = (float*)d_out;

  // workspace: f16 q [B][N][32] | k [B][N][32] | v [B][C][N]  (20 MB)
  _Float16* qf = (_Float16*)d_ws;
  _Float16* kf = qf + (size_t)BATCH * NPIX * DQK;
  _Float16* vf = kf + (size_t)BATCH * NPIX * DQK;

  dim3 pb(256);
  proj_kernel<true ><<<dim3(NPIX / 256, DQK, BATCH), pb, 0, stream>>>(Wq, bq, x, qf, DQK);
  proj_kernel<true ><<<dim3(NPIX / 256, DQK, BATCH), pb, 0, stream>>>(Wk, bk, x, kf, DQK);
  proj_kernel<false><<<dim3(NPIX / 256, CH,  BATCH), pb, 0, stream>>>(Wv, bv, x, vf, CH);

  attn_flash_wmma<<<dim3(NPIX / 16, BATCH), dim3(128), 0, stream>>>(
      qf, kf, vf, x, gamma, out);
}
